// XDeepFm_40879498728951
// MI455X (gfx1250) — matrix operand order repaired and checked by reading, not compiled, not run on hardware
//
#include <hip/hip_runtime.h>

// ---------------------------------------------------------------------------
// xDeepFM forward for MI455X (gfx1250): bf16 WMMA GEMMs for MLP + CIN,
// deterministic reductions, no atomics, all launches on `stream`.
// Round 3: MLP B-fragment software pipeline (no full loadcnt drain per WMMA),
// CIN outer-product build uses v_pk_mul_bf16 (packed bf16 multiply) to halve
// the VALU cost per Az dword.
// ---------------------------------------------------------------------------

#define NB 4096
#define FF 26
#define VV 100000
#define DD 16
#define HH 400
#define KK 128

typedef __attribute__((ext_vector_type(16))) __bf16 v16bf;
typedef __attribute__((ext_vector_type(8)))  float  v8f;

union AFrag { v16bf v; unsigned int u[8]; };

__device__ __forceinline__ unsigned short f2bf(float f) {
    unsigned u = __builtin_bit_cast(unsigned, f);
    return (unsigned short)((u + 0x7FFFu + ((u >> 16) & 1u)) >> 16);
}
__device__ __forceinline__ float bf2f(unsigned short h) {
    return __builtin_bit_cast(float, ((unsigned)h) << 16);
}
__device__ __forceinline__ v8f zero8() {
    v8f z;
#pragma unroll
    for (int i = 0; i < 8; ++i) z[i] = 0.f;
    return z;
}
__device__ __forceinline__ float waveRedSum(float v) {
#pragma unroll
    for (int off = 16; off; off >>= 1) v += __shfl_xor(v, off, 32);
    return v;
}
// packed bf16 multiply: d.lo = a.lo*b.lo, d.hi = a.hi*b.hi  (VOP3P op 42)
__device__ __forceinline__ unsigned pk_mul_bf16(unsigned a, unsigned b) {
    unsigned d;
    asm("v_pk_mul_bf16 %0, %1, %2" : "=v"(d) : "v"(a), "v"(b));
    return d;
}

// ------------------------------- zero fill ---------------------------------
__global__ void k_zero(unsigned int* p, int n) {
    for (int i = blockIdx.x * blockDim.x + threadIdx.x; i < n; i += gridDim.x * blockDim.x)
        p[i] = 0u;
}

// ------------------------- weight repack (bf16 B-frags) --------------------
// B fragment layout (32x16 bf16, K x N): lane = ncol + 16*(klocal>=16),
// halfword slot = klocal & 15.  Fragment = 32 lanes * 16 bf16, contiguous.
// CIN W: (128, Kdim) row-major, Kdim = L*26.  Bp[(kt*8+nt)*512 + lane*16 + pos]
__global__ void k_repack_cin(const float* __restrict__ W, unsigned short* __restrict__ Bp,
                             int Kdim, int total) {
    for (int id = blockIdx.x * blockDim.x + threadIdx.x; id < total;
         id += gridDim.x * blockDim.x) {
        int pos  = id & 15;
        int lane = (id >> 4) & 31;
        int nt   = (id >> 9) & 7;
        int kt   = id >> 12;
        int klocal = pos + ((lane >> 4) << 4);
        int k  = nt * 16 + (lane & 15);
        int lf = kt * 32 + klocal;
        float v = (lf < Kdim) ? W[(size_t)k * Kdim + lf] : 0.f;
        Bp[id] = f2bf(v);
    }
}

// MLP W: (Kin, 400) row-major.  13 K-tiles (pad to 416), 26 N-tiles (pad to 416).
__global__ void k_repack_mlp(const float* __restrict__ W, unsigned short* __restrict__ Bp,
                             int Kin) {
    const int total = 13 * 26 * 512;
    for (int id = blockIdx.x * blockDim.x + threadIdx.x; id < total;
         id += gridDim.x * blockDim.x) {
        int pos  = id & 15;
        int lane = (id >> 4) & 31;
        int rem  = id >> 9;
        int nt = rem % 26;
        int kt = rem / 26;
        int row = kt * 32 + pos + ((lane >> 4) << 4);
        int col = nt * 16 + (lane & 15);
        float v = (row < Kin && col < HH) ? W[(size_t)row * HH + col] : 0.f;
        Bp[id] = f2bf(v);
    }
}

// ------------------- gather: emb -> h0/x0, first-order + FM ----------------
__global__ void __launch_bounds__(256)
k_gather(const int* __restrict__ xidx, const float* __restrict__ emb,
         const float* __restrict__ w1t, const float* __restrict__ w1b,
         unsigned short* __restrict__ x0, unsigned short* __restrict__ h0,
         float* __restrict__ out) {
    int wv = threadIdx.x >> 5, lane = threadIdx.x & 31;
    int n = blockIdx.x * 8 + wv;
    const int* row = xidx + n * FF;
    float sd = 0.f, sq = 0.f;
    if (lane < 16) {
#pragma unroll 2
        for (int f = 0; f < FF; ++f) {
            int idx = row[f];
            float v = emb[(((size_t)f * VV + (unsigned)idx) << 4) + lane];
            sd += v; sq += v * v;
            unsigned short bv = f2bf(v);
            h0[(size_t)n * 416 + f * 16 + lane] = bv;
            x0[(((size_t)n << 4) + lane) * 32 + f] = bv;
        }
    }
    float t1 = sd * sd, t2 = sq;
#pragma unroll
    for (int off = 8; off; off >>= 1) {
        t1 += __shfl_xor(t1, off, 32);
        t2 += __shfl_xor(t2, off, 32);
    }
    float lin = 0.f;
    if (lane < FF) lin = w1t[(size_t)lane * VV + (unsigned)row[lane]];
    lin = waveRedSum(lin);
    if (lane == 0) out[n] += lin + w1b[0] + 0.5f * (t1 - t2);
}

// --------------------------- MLP GEMM (bf16 WMMA) --------------------------
// A: (4096 x 416) bf16 row-major, Out: (4096 x 416) bf16.
// 8 waves = 4 row-subtiles x 2 col-groups; each col-group owns 13 of 26
// n-tiles (tile 25 is the zero pad -> writes exact zeros to cols 400..415).
// B fragments are software-pipelined one deep: the wait before each WMMA
// tolerates the next fragment's in-flight load instead of draining loadcnt.
__global__ void __launch_bounds__(256)
k_mlp_gemm(const unsigned short* __restrict__ A, const unsigned short* __restrict__ Bp,
           const float* __restrict__ bias, unsigned short* __restrict__ Out, int relu) {
    __shared__ unsigned int As32[64 * 212];   // 64 rows x 424 halfwords (padded)
    int tid = threadIdx.x;
    size_t rowBase = (size_t)blockIdx.x * 64;
    const unsigned int* src = (const unsigned int*)A;
    for (int i = tid; i < 64 * 208; i += 256) {
        int r = i / 208, c = i - r * 208;
        As32[r * 212 + c] = src[(rowBase + r) * 208 + c];
    }
    __syncthreads();

    int wv = tid >> 5, lane = tid & 31;
    int rowsub   = wv & 3;
    int colgroup = wv >> 2;
    int ntBegin  = colgroup * 13;

    v8f acc[13];
#pragma unroll
    for (int j = 0; j < 13; ++j) acc[j] = zero8();

    const unsigned short* Ash = (const unsigned short*)As32;
    int mm = (rowsub << 4) + (lane & 15);
    int kb = (lane & 16) ? 16 : 0;

    for (int kt = 0; kt < 13; ++kt) {
        AFrag af;
        *(uint4*)&af.u[0] = *(const uint4*)(Ash + mm * 424 + kt * 32 + kb);
        *(uint4*)&af.u[4] = *(const uint4*)(Ash + mm * 424 + kt * 32 + kb + 8);
        const unsigned short* bbase =
            Bp + (((size_t)kt * 26 + ntBegin) << 9) + (lane << 4);
        v16bf bcur = *(const v16bf*)(bbase);
#pragma unroll
        for (int j = 0; j < 13; ++j) {
            v16bf bnext;
            if (j < 12) bnext = *(const v16bf*)(bbase + (size_t)(j + 1) * 512);
            acc[j] = __builtin_amdgcn_wmma_f32_16x16x32_bf16(
                false, af.v, false, bcur, (short)0, acc[j], false, false);
            bcur = bnext;
        }
    }

    int rowHi = (lane & 16) ? 8 : 0;
    size_t ob = rowBase + rowsub * 16 + rowHi;
#pragma unroll
    for (int j = 0; j < 13; ++j) {
        int nt  = ntBegin + j;
        int col = nt * 16 + (lane & 15);
        float b = (col < HH) ? bias[col] : 0.f;
#pragma unroll
        for (int r = 0; r < 8; ++r) {
            float v = acc[j][r] + b;
            if (relu && v < 0.f) v = 0.f;
            Out[(ob + r) * 416 + col] = f2bf(v);
        }
    }
}

// ------------------------- MLP final dot (400 -> 1) ------------------------
__global__ void __launch_bounds__(256)
k_mlp_dot(const unsigned short* __restrict__ h2, const float* __restrict__ w3,
          const float* __restrict__ b3, float* __restrict__ out) {
    int wv = threadIdx.x >> 5, lane = threadIdx.x & 31;
    int n = blockIdx.x * 8 + wv;
    float s = 0.f;
    for (int c = lane; c < HH; c += 32) s += bf2f(h2[(size_t)n * 416 + c]) * w3[c];
    s = waveRedSum(s);
    if (lane == 0) out[n] += s + b3[0];
}

// ---------------- CIN layer: outer-product-on-the-fly WMMA GEMM ------------
// out[m,k] = sum_lf W[k,lf] * last[m, lf/26] * x0[m, lf%26],  m = n*16+d.
// Double-buffered Az tile: one barrier per K-step.  B-fragment loads issued
// before the VALU build phase so global latency overlaps the outer-product.
// Az dwords built with v_pk_mul_bf16 (two bf16 products per instruction).
__global__ void __launch_bounds__(256)
k_cin(const unsigned short* __restrict__ last, const unsigned short* __restrict__ x0,
      const unsigned short* __restrict__ Bp, const float* __restrict__ linw,
      unsigned short* __restrict__ outLast, float* __restrict__ cinPart,
      int partOff, int Lw, int Kdim, int KT) {
    __shared__ unsigned int lastT[64 * 64];     // 64 rows x Lw bf16
    __shared__ unsigned int x0T[64 * 16];       // 64 rows x 32 bf16
    __shared__ unsigned int Az32[2][64 * 20];   // 2 x (64 rows x 32 bf16, stride 40h)
    __shared__ unsigned int tbl[3328];          // lf -> (l<<16)|f

    int tid = threadIdx.x;
    size_t mBase = (size_t)blockIdx.x * 64;
    int Lw2 = Lw >> 1;

    const unsigned int* lsrc = (const unsigned int*)last;
    for (int i = tid; i < 64 * Lw2; i += 256) {
        int r = i / Lw2, c = i - r * Lw2;
        lastT[r * Lw2 + c] = lsrc[(mBase + r) * Lw2 + c];
    }
    const unsigned int* xsrc = (const unsigned int*)x0;
    for (int i = tid; i < 64 * 16; i += 256) {
        int r = i >> 4, c = i & 15;
        x0T[r * 16 + c] = xsrc[(mBase + r) * 16 + c];
    }
    int Ktot = KT * 32;
    for (int lf = tid; lf < Ktot; lf += 256) {
        int l = 0, f = 0;
        if (lf < Kdim) { l = lf / 26; f = lf - l * 26; }
        tbl[lf] = ((unsigned)l << 16) | (unsigned)f;
    }
    __syncthreads();

    int wv = tid >> 5, lane = tid & 31;
    int rowsub = wv & 3, colgroup = wv >> 2;

    v8f acc[4];
#pragma unroll
    for (int t = 0; t < 4; ++t) acc[t] = zero8();

    const unsigned short* lastTh = (const unsigned short*)lastT;
    const unsigned short* x0Th   = (const unsigned short*)x0T;
    int mm = (rowsub << 4) + (lane & 15);
    int kb = (lane & 16) ? 16 : 0;

    for (int kt = 0; kt < KT; ++kt) {
        int buf = kt & 1;
        // ---- B fragments for this K-step: independent of LDS, issue early
        const unsigned short* bbase =
            Bp + (((size_t)kt * 8 + (colgroup << 2)) << 9) + (lane << 4);
        v16bf bv0 = *(const v16bf*)(bbase);
        v16bf bv1 = *(const v16bf*)(bbase + 512);
        v16bf bv2 = *(const v16bf*)(bbase + 1024);
        v16bf bv3 = *(const v16bf*)(bbase + 1536);

        // ---- cooperative build of 64x32 outer-product tile (overlaps loads)
        unsigned int* AzW = Az32[buf];
        int lfBase = kt << 5;
#pragma unroll
        for (int e = 0; e < 4; ++e) {
            int i = e * 256 + tid;            // 1024 dwords total
            int m = i >> 4, c = i & 15;
            const unsigned short* lrow = lastTh + m * Lw;
            const unsigned short* xrow = x0Th + (m << 5);
            unsigned t0 = tbl[lfBase + (c << 1)];
            unsigned t1 = tbl[lfBase + (c << 1) + 1];
            unsigned a = (unsigned)lrow[t0 >> 16] | ((unsigned)lrow[t1 >> 16] << 16);
            unsigned b = (unsigned)xrow[t0 & 0xffffu] | ((unsigned)xrow[t1 & 0xffffu] << 16);
            AzW[m * 20 + c] = pk_mul_bf16(a, b);
        }
        __syncthreads();   // single barrier per step (double-buffered Az)

        const unsigned short* AzR = (const unsigned short*)Az32[buf];
        AFrag af;
        *(uint4*)&af.u[0] = *(const uint4*)(AzR + mm * 40 + kb);
        *(uint4*)&af.u[4] = *(const uint4*)(AzR + mm * 40 + kb + 8);

        acc[0] = __builtin_amdgcn_wmma_f32_16x16x32_bf16(
            false, af.v, false, bv0, (short)0, acc[0], false, false);
        acc[1] = __builtin_amdgcn_wmma_f32_16x16x32_bf16(
            false, af.v, false, bv1, (short)0, acc[1], false, false);
        acc[2] = __builtin_amdgcn_wmma_f32_16x16x32_bf16(
            false, af.v, false, bv2, (short)0, acc[2], false, false);
        acc[3] = __builtin_amdgcn_wmma_f32_16x16x32_bf16(
            false, af.v, false, bv3, (short)0, acc[3], false, false);
    }

    // Epilogue: store bf16 'last' for next layer; reduce lin_w-weighted sum
    // over this sample's 16 d-rows and each n-tile's 16 cols; one partial per n-tile (deterministic).
    int n = blockIdx.x * 4 + rowsub;
    int rowHi = (lane & 16) ? 8 : 0;
    size_t ob = mBase + rowsub * 16 + rowHi;
#pragma unroll
    for (int t = 0; t < 4; ++t) {
        int nt  = (colgroup << 2) + t;
        int col = nt * 16 + (lane & 15);
        float lw = linw[col];
        float ps = 0.f;
#pragma unroll
        for (int r = 0; r < 8; ++r) {
            float v = acc[t][r];
            outLast[(ob + r) * 128 + col] = f2bf(v);
            ps += v;
        }
        float tot = waveRedSum(ps * lw);
        if (lane == 0) cinPart[n * 24 + partOff + nt] = tot;
    }
}

// ------------------------------ final combine ------------------------------
__global__ void k_combine(const float* __restrict__ cinPart, const float* __restrict__ lb,
                          float* __restrict__ out) {
    int n = blockIdx.x * 256 + threadIdx.x;
    if (n < NB) {
        float s = lb[0];
#pragma unroll
        for (int i = 0; i < 24; ++i) s += cinPart[n * 24 + i];
        out[n] += s;
    }
}

// ---------------------------------------------------------------------------
extern "C" void kernel_launch(void* const* d_in, const int* in_sizes, int n_in,
                              void* d_out, int out_size, void* d_ws, size_t ws_size,
                              hipStream_t stream) {
    const int*   x_idx = (const int*)  d_in[0];
    const float* embed = (const float*)d_in[1];
    const float* w1t   = (const float*)d_in[2];
    const float* w1b   = (const float*)d_in[3];
    const float* mw1   = (const float*)d_in[4];
    const float* mb1   = (const float*)d_in[5];
    const float* mw2   = (const float*)d_in[6];
    const float* mb2   = (const float*)d_in[7];
    const float* mw3   = (const float*)d_in[8];
    const float* mb3   = (const float*)d_in[9];
    const float* cw0   = (const float*)d_in[10];
    const float* cw1   = (const float*)d_in[11];
    const float* cw2   = (const float*)d_in[12];
    const float* lw    = (const float*)d_in[13];
    const float* lb    = (const float*)d_in[14];
    float* out = (float*)d_out;

    char* ws = (char*)d_ws;
    size_t off = 0;
    auto alloc = [&](size_t bytes) -> void* {
        void* p = ws + off;
        off += (bytes + 255) & ~(size_t)255;
        return p;
    };
    unsigned short* x0    = (unsigned short*)alloc((size_t)65536 * 32 * 2);   // 4 MB
    unsigned short* h0    = (unsigned short*)alloc((size_t)NB * 416 * 2);
    unsigned short* h1    = (unsigned short*)alloc((size_t)NB * 416 * 2);
    unsigned short* h2    = (unsigned short*)alloc((size_t)NB * 416 * 2);
    unsigned short* last0 = (unsigned short*)alloc((size_t)65536 * 128 * 2);  // 16 MB
    unsigned short* last1 = (unsigned short*)alloc((size_t)65536 * 128 * 2);
    unsigned short* Bm1   = (unsigned short*)alloc((size_t)13 * 26 * 512 * 2);
    unsigned short* Bm2   = (unsigned short*)alloc((size_t)13 * 26 * 512 * 2);
    unsigned short* Bc0   = (unsigned short*)alloc((size_t)22 * 8 * 512 * 2);
    unsigned short* Bc1   = (unsigned short*)alloc((size_t)104 * 8 * 512 * 2);
    unsigned short* Bc2   = (unsigned short*)alloc((size_t)104 * 8 * 512 * 2);
    float*          cpart = (float*)alloc((size_t)NB * 24 * 4);
    if (ws_size < off) return;  // workspace too small; bail safely

    // 1) zero init: d_out and x0 (pad columns of x0 must read as 0.0)
    k_zero<<<1024, 256, 0, stream>>>((unsigned int*)x0, 65536 * 32 * 2 / 4);
    k_zero<<<16,   256, 0, stream>>>((unsigned int*)out, NB);

    // 2) weight repacks (recomputed every call -> deterministic, cheap)
    k_repack_mlp<<<676, 256, 0, stream>>>(mw1, Bm1, 416);
    k_repack_mlp<<<676, 256, 0, stream>>>(mw2, Bm2, 400);
    k_repack_cin<<<352,  256, 0, stream>>>(cw0, Bc0, 676,  22 * 8 * 512);
    k_repack_cin<<<1664, 256, 0, stream>>>(cw1, Bc1, 3328, 104 * 8 * 512);
    k_repack_cin<<<1664, 256, 0, stream>>>(cw2, Bc2, 3328, 104 * 8 * 512);

    // 3) gather + first-order + FM
    k_gather<<<NB / 8, 256, 0, stream>>>(x_idx, embed, w1t, w1b, x0, h0, out);

    // 4) MLP (gemm writes pad cols 400..415 as exact zeros via padded B tile)
    k_mlp_gemm<<<NB / 64, 256, 0, stream>>>(h0, Bm1, mb1, h1, 1);
    k_mlp_gemm<<<NB / 64, 256, 0, stream>>>(h1, Bm2, mb2, h2, 1);
    k_mlp_dot<<<NB / 8, 256, 0, stream>>>(h2, mw3, mb3, out);

    // 5) CIN layers (last ping-pong: x0 -> last0 -> last1 -> last0)
    k_cin<<<1024, 256, 0, stream>>>(x0,    x0, Bc0, lw,        last0, cpart, 0,  32,  676,  22);
    k_cin<<<1024, 256, 0, stream>>>(last0, x0, Bc1, lw + 128,  last1, cpart, 8,  128, 3328, 104);
    k_cin<<<1024, 256, 0, stream>>>(last1, x0, Bc2, lw + 256,  last0, cpart, 16, 128, 3328, 104);

    // 6) combine CIN partials + lin_b into d_out
    k_combine<<<16, 256, 0, stream>>>(cpart, lb, out);
}
